// CausalSelfAttention_85633057948175
// MI455X (gfx1250) — matrix operand order repaired and checked by reading, not compiled, not run on hardware
//
#include <hip/hip_runtime.h>
#include <hip/hip_bf16.h>

typedef __attribute__((ext_vector_type(16))) _Float16 v16h;
typedef __attribute__((ext_vector_type(8)))  float    v8f;
typedef __attribute__((ext_vector_type(4)))  unsigned int v4u;
typedef __attribute__((ext_vector_type(4)))  float    v4f;

union V16H { v16h v; v4u q[2]; };
union PK8  { _Float16 h[8]; v4u u; };

#define WMMA_F32_F16(a, b, c) \
    __builtin_amdgcn_wmma_f32_16x16x32_f16(false, (a), false, (b), (short)0, (c), false, false)

// --- gfx1250 async global->LDS copy (ASYNCcnt-tracked), per-lane 16B ---------
__device__ __forceinline__ void async_copy_b128(const void* gptr, void* lptr) {
    unsigned lds = (unsigned)(size_t)lptr;   // flat->LDS: low 32 bits are LDS offset
    asm volatile("global_load_async_to_lds_b128 %0, %1, off"
                 :: "v"(lds), "v"(gptr) : "memory");
}
__device__ __forceinline__ void wait_async0() {
    asm volatile("s_wait_asynccnt 0x0" ::: "memory");
}

// ---------------------------------------------------------------------------
// Projection GEMM: Y = X(f32)[8192,1024] @ W(f32)[1024,1024] + bias
// mode 0: store f16 as [b][h][s][d]   (Q, K)
// mode 1: store f16 as [b][h][d][s]   (V transposed, for contiguous PV B-frags)
// Block tile 256x64, 8 waves, each wave 32 rows x 64 cols (2 A-frags reuse
// the 4 B-frags: 8 WMMAs per 12 ds_load_b128), K-step 32.
// ---------------------------------------------------------------------------
__global__ __launch_bounds__(256) void gemm_qkv_kernel(
    const float* __restrict__ X, const float* __restrict__ W,
    const float* __restrict__ bias, _Float16* __restrict__ out, int mode)
{
    __shared__ _Float16 Xs[256][32];  // [m][k] f16, 16 KB
    __shared__ _Float16 Ws[64][32];   // [n][k] f16 (k-contiguous), 4 KB

    const int t    = threadIdx.x;
    const int lane = t & 31;
    const int wv   = t >> 5;
    const int m0   = blockIdx.x * 256;
    const int n0   = blockIdx.y * 64;
    const int K = 1024, N = 1024;

    const int l15  = lane & 15;
    const int off8 = (lane < 16) ? 0 : 8;
    const int kb16 = (lane < 16) ? 0 : 16;

    const int wn = t & 63, wkg = (t >> 6) * 8;   // W: thread owns col n, 8 k-rows

    v8f acc[2][4] = {};

    for (int kk = 0; kk < K; kk += 32) {
        __syncthreads();
        // --- X tile: one row per thread, f32 -> f16, 4 packed 16B LDS stores
        {
            const v4f* src = (const v4f*)(X + (size_t)(m0 + t) * K + kk);
            #pragma unroll
            for (int g = 0; g < 4; ++g) {
                v4f f0 = src[2 * g], f1 = src[2 * g + 1];
                PK8 p;
                p.h[0]=(_Float16)f0.x; p.h[1]=(_Float16)f0.y; p.h[2]=(_Float16)f0.z; p.h[3]=(_Float16)f0.w;
                p.h[4]=(_Float16)f1.x; p.h[5]=(_Float16)f1.y; p.h[6]=(_Float16)f1.z; p.h[7]=(_Float16)f1.w;
                *(v4u*)&Xs[t][g * 8] = p.u;
            }
        }
        // --- W tile: thread owns one column n (lane-coalesced loads),
        //     8 strided k-rows -> one k-contiguous ds_store_b128
        {
            const float* wc = W + (size_t)(kk + wkg) * N + n0 + wn;
            PK8 p;
            #pragma unroll
            for (int i = 0; i < 8; ++i) p.h[i] = (_Float16)wc[(size_t)i * N];
            *(v4u*)&Ws[wn][wkg] = p.u;
        }
        if (kk + 32 < K)
            __builtin_prefetch(W + (size_t)(kk + 32) * N + n0 + wn, 0, 1);
        __syncthreads();

        // --- fragments: 2 A frags + 4 B frags, then 8 back-to-back WMMAs
        const int mrow = wv * 32 + l15;
        V16H ua[2], ub[4];
        #pragma unroll
        for (int mt = 0; mt < 2; ++mt) {
            ua[mt].q[0] = *(const v4u*)&Xs[mrow + mt * 16][kb16];
            ua[mt].q[1] = *(const v4u*)&Xs[mrow + mt * 16][kb16 + 8];
        }
        #pragma unroll
        for (int nt = 0; nt < 4; ++nt) {
            const int ncol = nt * 16 + l15;
            ub[nt].q[0] = *(const v4u*)&Ws[ncol][kb16];
            ub[nt].q[1] = *(const v4u*)&Ws[ncol][kb16 + 8];
        }
        #pragma unroll
        for (int mt = 0; mt < 2; ++mt)
            #pragma unroll
            for (int nt = 0; nt < 4; ++nt)
                acc[mt][nt] = WMMA_F32_F16(ua[mt].v, ub[nt].v, acc[mt][nt]);
    }

    // --- epilogue: bias + scatter to head layout
    #pragma unroll
    for (int nt = 0; nt < 4; ++nt) {
        const int n  = n0 + nt * 16 + l15;
        const float bv = bias[n];
        const int h = n >> 6, dd = n & 63;
        #pragma unroll
        for (int mt = 0; mt < 2; ++mt) {
            #pragma unroll
            for (int r = 0; r < 8; ++r) {
                const int m = m0 + wv * 32 + mt * 16 + r + ((lane < 16) ? 0 : 8);
                const int b = m >> 11, s = m & 2047;
                const float y = acc[mt][nt][r] + bv;
                size_t idx;
                if (mode == 0) idx = ((size_t)((b * 16 + h) * 2048 + s)) * 64 + dd;
                else           idx = ((size_t)((b * 16 + h) * 64 + dd)) * 2048 + s;
                out[idx] = (_Float16)y;
            }
        }
    }
}

// ---------------------------------------------------------------------------
// Flash attention, causal. One wave per 16-query tile, 32 keys per step.
// q16,k16: [b][h][s][64] f16.  vt16: [b][h][64][s] f16.
// att16 out: [b][s][h*64+d] f16 (row-major for the output GEMM).
// ---------------------------------------------------------------------------
__global__ __launch_bounds__(256) void attn_kernel(
    const _Float16* __restrict__ q16, const _Float16* __restrict__ k16,
    const _Float16* __restrict__ vt16, _Float16* __restrict__ att16)
{
    __shared__ _Float16 Plds[8][16][32];  // per-wave P staging, 8 KB

    const int t = threadIdx.x, lane = t & 31, wv = t >> 5;
    const int b = blockIdx.z, h = blockIdx.y;
    const int qtile = blockIdx.x * 8 + wv;
    const int q0 = qtile * 16;

    const size_t qkbase = ((size_t)(b * 16 + h)) * 2048 * 64;
    const size_t vtbase = ((size_t)(b * 16 + h)) * 64 * 2048;
    const int l15  = lane & 15;
    const int off8 = (lane < 16) ? 0 : 8;
    const int kb16 = (lane < 16) ? 0 : 16;
    const int rbase = (lane < 16) ? 0 : 8;  // C-layout row base for this half

    // Q A-fragments for the two 32-wide d-steps (loaded once)
    V16H qa[2];
    {
        const _Float16* qrow = q16 + qkbase + (size_t)(q0 + l15) * 64;
        #pragma unroll
        for (int ks = 0; ks < 2; ++ks) {
            qa[ks].q[0] = *(const v4u*)(qrow + ks * 32 + kb16);
            qa[ks].q[1] = *(const v4u*)(qrow + ks * 32 + kb16 + 8);
        }
    }

    v8f acc[4] = {};
    float rmax[8], rsum[8];
    #pragma unroll
    for (int r = 0; r < 8; ++r) { rmax[r] = -1e30f; rsum[r] = 0.f; }
    const float sscale = 0.125f;  // 1/sqrt(64)

    for (int kv0 = 0; kv0 <= q0 + 15; kv0 += 32) {
        // ---- S = Q @ K^T for 32 keys (2 col-tiles x 2 d-steps)
        v8f S[2] = {};
        #pragma unroll
        for (int ct = 0; ct < 2; ++ct) {
            const int key = kv0 + ct * 16 + l15;
            const _Float16* krow = k16 + qkbase + (size_t)key * 64;
            V16H kf0, kf1;
            kf0.q[0] = *(const v4u*)(krow + kb16);
            kf0.q[1] = *(const v4u*)(krow + kb16 + 8);
            kf1.q[0] = *(const v4u*)(krow + 32 + kb16);
            kf1.q[1] = *(const v4u*)(krow + 32 + kb16 + 8);
            S[ct] = WMMA_F32_F16(qa[0].v, kf0.v, S[ct]);
            S[ct] = WMMA_F32_F16(qa[1].v, kf1.v, S[ct]);
        }

        // ---- causal mask + scale + tile row-max
        float tmax[8];
        #pragma unroll
        for (int r = 0; r < 8; ++r) tmax[r] = -1e30f;
        #pragma unroll
        for (int ct = 0; ct < 2; ++ct) {
            const int key = kv0 + ct * 16 + l15;
            #pragma unroll
            for (int r = 0; r < 8; ++r) {
                float s = S[ct][r] * sscale;
                s = (key <= q0 + rbase + r) ? s : -1e30f;
                S[ct][r] = s;
                tmax[r] = fmaxf(tmax[r], s);
            }
        }
        #pragma unroll
        for (int m = 1; m < 16; m <<= 1)
            #pragma unroll
            for (int r = 0; r < 8; ++r)
                tmax[r] = fmaxf(tmax[r], __shfl_xor(tmax[r], m, 32));

        // ---- online softmax update
        float corr[8], psum[8];
        #pragma unroll
        for (int r = 0; r < 8; ++r) {
            const float nm = fmaxf(rmax[r], tmax[r]);
            corr[r] = __expf(rmax[r] - nm);
            rmax[r] = nm;
            psum[r] = 0.f;
        }
        #pragma unroll
        for (int ct = 0; ct < 2; ++ct) {
            #pragma unroll
            for (int r = 0; r < 8; ++r) {
                const float p = __expf(S[ct][r] - rmax[r]);
                psum[r] += p;
                Plds[wv][rbase + r][ct * 16 + l15] = (_Float16)p;
            }
        }
        #pragma unroll
        for (int m = 1; m < 16; m <<= 1)
            #pragma unroll
            for (int r = 0; r < 8; ++r)
                psum[r] += __shfl_xor(psum[r], m, 32);
        #pragma unroll
        for (int r = 0; r < 8; ++r) rsum[r] = rsum[r] * corr[r] + psum[r];
        #pragma unroll
        for (int nt = 0; nt < 4; ++nt)
            #pragma unroll
            for (int r = 0; r < 8; ++r) acc[nt][r] *= corr[r];

        // ---- acc += P @ V  (P via LDS C->A layout conversion; same wave, DS in-order)
        V16H pa, vb[4];
        {
            const _Float16* prow = &Plds[wv][l15][0];
            pa.q[0] = *(const v4u*)(prow + kb16);
            pa.q[1] = *(const v4u*)(prow + kb16 + 8);
        }
        #pragma unroll
        for (int nt = 0; nt < 4; ++nt) {
            const int d = nt * 16 + l15;
            const _Float16* vp = vt16 + vtbase + (size_t)d * 2048 + kv0 + kb16;
            vb[nt].q[0] = *(const v4u*)vp;
            vb[nt].q[1] = *(const v4u*)(vp + 8);
        }
        #pragma unroll
        for (int nt = 0; nt < 4; ++nt)
            acc[nt] = WMMA_F32_F16(pa.v, vb[nt].v, acc[nt]);
    }

    // ---- normalize + store to [b][s][h*64+d]
    #pragma unroll
    for (int nt = 0; nt < 4; ++nt) {
        const int d = h * 64 + nt * 16 + l15;
        #pragma unroll
        for (int r = 0; r < 8; ++r) {
            const int s = q0 + rbase + r;
            const float y = acc[nt][r] / rsum[r];
            att16[((size_t)(b * 2048 + s)) * 1024 + d] = (_Float16)y;
        }
    }
}

// ---------------------------------------------------------------------------
// Output GEMM: out(f32) = att16(f16)[8192,1024] @ Wo(f32) + bo
// X tile copied with GLOBAL_LOAD_ASYNC_TO_LDS_B128 (ASYNCcnt path).
// Block tile 256x64, each wave 32x64 (8 WMMAs per k-step).
// ---------------------------------------------------------------------------
__global__ __launch_bounds__(256) void gemm_out_kernel(
    const _Float16* __restrict__ Xh, const float* __restrict__ W,
    const float* __restrict__ bias, float* __restrict__ out)
{
    __shared__ _Float16 Xs[256][32];
    __shared__ _Float16 Ws[64][32];

    const int t    = threadIdx.x;
    const int lane = t & 31;
    const int wv   = t >> 5;
    const int m0   = blockIdx.x * 256;
    const int n0   = blockIdx.y * 64;
    const int K = 1024, N = 1024;

    const int l15  = lane & 15;
    const int off8 = (lane < 16) ? 0 : 8;
    const int kb16 = (lane < 16) ? 0 : 16;

    const int wn = t & 63, wkg = (t >> 6) * 8;

    v8f acc[2][4] = {};

    for (int kk = 0; kk < K; kk += 32) {
        __syncthreads();
        {   // X already f16: async global->LDS, one row (4x16B) per thread
            const _Float16* src = Xh + (size_t)(m0 + t) * K + kk;
            #pragma unroll
            for (int g = 0; g < 4; ++g)
                async_copy_b128(src + g * 8, &Xs[t][g * 8]);
        }
        {   // W tile: column-owned, packed k-contiguous store
            const float* wc = W + (size_t)(kk + wkg) * N + n0 + wn;
            PK8 p;
            #pragma unroll
            for (int i = 0; i < 8; ++i) p.h[i] = (_Float16)wc[(size_t)i * N];
            *(v4u*)&Ws[wn][wkg] = p.u;
        }
        if (kk + 32 < K)
            __builtin_prefetch(W + (size_t)(kk + 32) * N + n0 + wn, 0, 1);
        wait_async0();          // this wave's async LDS writes done
        __syncthreads();        // everyone's writes visible

        const int mrow = wv * 32 + l15;
        V16H ua[2], ub[4];
        #pragma unroll
        for (int mt = 0; mt < 2; ++mt) {
            ua[mt].q[0] = *(const v4u*)&Xs[mrow + mt * 16][kb16];
            ua[mt].q[1] = *(const v4u*)&Xs[mrow + mt * 16][kb16 + 8];
        }
        #pragma unroll
        for (int nt = 0; nt < 4; ++nt) {
            const int ncol = nt * 16 + l15;
            ub[nt].q[0] = *(const v4u*)&Ws[ncol][kb16];
            ub[nt].q[1] = *(const v4u*)&Ws[ncol][kb16 + 8];
        }
        #pragma unroll
        for (int mt = 0; mt < 2; ++mt)
            #pragma unroll
            for (int nt = 0; nt < 4; ++nt)
                acc[mt][nt] = WMMA_F32_F16(ua[mt].v, ub[nt].v, acc[mt][nt]);
    }

    #pragma unroll
    for (int nt = 0; nt < 4; ++nt) {
        const int n  = n0 + nt * 16 + l15;
        const float bv = bias[n];
        #pragma unroll
        for (int mt = 0; mt < 2; ++mt) {
            #pragma unroll
            for (int r = 0; r < 8; ++r) {
                const int m = m0 + wv * 32 + mt * 16 + r + ((lane < 16) ? 0 : 8);
                out[(size_t)m * 1024 + n] = acc[mt][nt][r] + bv;
            }
        }
    }
}

// ---------------------------------------------------------------------------
extern "C" void kernel_launch(void* const* d_in, const int* in_sizes, int n_in,
                              void* d_out, int out_size, void* d_ws, size_t ws_size,
                              hipStream_t stream) {
    (void)in_sizes; (void)n_in; (void)out_size; (void)ws_size;
    const float* x  = (const float*)d_in[0];
    const float* wq = (const float*)d_in[1];
    const float* bq = (const float*)d_in[2];
    const float* wk = (const float*)d_in[3];
    const float* bk = (const float*)d_in[4];
    const float* wv = (const float*)d_in[5];
    const float* bv = (const float*)d_in[6];
    const float* wo = (const float*)d_in[7];
    const float* bo = (const float*)d_in[8];
    float* out = (float*)d_out;

    const size_t NE = (size_t)8192 * 1024;
    _Float16* ws    = (_Float16*)d_ws;
    _Float16* q16   = ws;
    _Float16* k16   = ws + NE;
    _Float16* vt16  = ws + 2 * NE;
    _Float16* att16 = ws + 3 * NE;   // 64 MB total f16 workspace

    dim3 blk(256);
    dim3 ggemm(32, 16, 1);           // 8192/256 x 1024/64
    gemm_qkv_kernel<<<ggemm, blk, 0, stream>>>(x, wq, bq, q16, 0);
    gemm_qkv_kernel<<<ggemm, blk, 0, stream>>>(x, wk, bk, k16, 0);
    gemm_qkv_kernel<<<ggemm, blk, 0, stream>>>(x, wv, bv, vt16, 1);

    dim3 gatt(16, 16, 4);            // qtile-chunks x heads x batch
    attn_kernel<<<gatt, blk, 0, stream>>>(q16, k16, vt16, att16);

    gemm_out_kernel<<<ggemm, blk, 0, stream>>>(att16, wo, bo, out);
}
